// CausalAttentionLayer_67465346286176
// MI455X (gfx1250) — matrix operand order repaired and checked by reading, not compiled, not run on hardware
//
#include <hip/hip_runtime.h>

// ---------------- CDNA5 WMMA / TDM types ----------------
typedef __attribute__((ext_vector_type(16))) __bf16 v16bf;
typedef __attribute__((ext_vector_type(8)))  float  v8f;
typedef __attribute__((ext_vector_type(4)))  unsigned int v4u;
typedef __attribute__((ext_vector_type(8)))  int v8i;
typedef __attribute__((ext_vector_type(4)))  int v4i;

#if __has_builtin(__builtin_amdgcn_tensor_load_to_lds)
#define USE_TDM 1
#else
#define USE_TDM 0
#endif

// Problem constants (B,N,C,H,D fixed by the reference)
#define BD 4
#define ND 2048
#define CD 1024
#define HD 16
#define DD 64
#define KD 1024      // reduction dim of both dense GEMMs (= C)

union V16U {
    v16bf v;
    uint4 q[2];
    unsigned short s[16];
};

__device__ __forceinline__ unsigned short f2bf(float f) {
    unsigned u = __float_as_uint(f);
    u += 0x7FFFu + ((u >> 16) & 1u);   // round-to-nearest-even
    return (unsigned short)(u >> 16);
}

__device__ __forceinline__ v8f v8f_zero() {
    v8f z;
#pragma unroll
    for (int i = 0; i < 8; ++i) z[i] = 0.0f;
    return z;
}

// ---------------- fp32 -> bf16 conversion ----------------
__global__ __launch_bounds__(256) void cvt_f32_bf16(const float* __restrict__ in,
                                                    unsigned short* __restrict__ out, int n) {
    int i = blockIdx.x * 256 + threadIdx.x;
    if (i < n) out[i] = f2bf(in[i]);
}

// =====================================================================
// Dense GEMM core: per-wave 32(M) x 64(N) tile, K-step 32, register
// double buffering so next k-step's loads overlap current WMMAs.
// A-layout:  lane m=lane&15 owns row, K(e) = (e<8?e:e+8)+8*(lane>>4)
// B-layout:  lane owns column (= weight row), K(e) = (e<8?e:e+8)+8*(lane>>4), same as A
// =====================================================================

// ---------------- QKV GEMM: [8192,1024] x [3072,1024]^T -> Q/K/V [B,H,N,D] bf16 ----------------
__global__ __launch_bounds__(256) void qkv_gemm(const unsigned short* __restrict__ Xb,
                                                const unsigned short* __restrict__ Wb,
                                                unsigned short* __restrict__ Q,
                                                unsigned short* __restrict__ Kt,
                                                unsigned short* __restrict__ Vt) {
    const int lane = threadIdx.x & 31;
    const int widx = threadIdx.x >> 5;
    const int wg   = blockIdx.x * 8 + widx;
    const int mtiles = (BD * ND) / 32;            // 256
    const int tm   = (wg % mtiles) * 32;          // row base
    const int tn   = (wg / mtiles) * 64;          // col base (0..3008)
    const int half = lane >> 4;
    const int ml   = lane & 15;

    const unsigned short* arow[2];
    arow[0] = Xb + (long)(tm + ml) * KD;
    arow[1] = Xb + (long)(tm + 16 + ml) * KD;
    const unsigned short* brow[4];
#pragma unroll
    for (int j = 0; j < 4; ++j) brow[j] = Wb + (long)(tn + j * 16 + ml) * KD;

    v8f acc[2][4];
#pragma unroll
    for (int mi = 0; mi < 2; ++mi)
#pragma unroll
        for (int j = 0; j < 4; ++j) acc[mi][j] = v8f_zero();

    V16U a[2][2], b[2][4];                        // [buf][tile]
    auto loadA = [&](int buf, int k0) {
#pragma unroll
        for (int mi = 0; mi < 2; ++mi) {
            const unsigned short* p = arow[mi] + k0 + 8 * half;
            a[buf][mi].q[0] = *(const uint4*)(p);
            a[buf][mi].q[1] = *(const uint4*)(p + 16);
        }
    };
    auto loadB = [&](int buf, int k0) {
#pragma unroll
        for (int j = 0; j < 4; ++j) {
            const unsigned short* p = brow[j] + k0 + 8 * half;
            b[buf][j].q[0] = *(const uint4*)(p);
            b[buf][j].q[1] = *(const uint4*)(p + 16);
        }
    };

    loadA(0, 0);
    loadB(0, 0);
    for (int k0 = 0; k0 < KD; k0 += 32) {
        const int cur = (k0 >> 5) & 1;
        const int nxt = cur ^ 1;
        if (k0 + 32 < KD) { loadA(nxt, k0 + 32); loadB(nxt, k0 + 32); }
#pragma unroll
        for (int mi = 0; mi < 2; ++mi)
#pragma unroll
            for (int j = 0; j < 4; ++j)
                acc[mi][j] = __builtin_amdgcn_wmma_f32_16x16x32_bf16(
                    false, a[cur][mi].v, false, b[cur][j].v, (short)0, acc[mi][j], false, false);
    }

    // Scatter C-layout tiles into Q/K/V [B,H,N,D] bf16
    unsigned short* dsts[3] = {Q, Kt, Vt};
#pragma unroll
    for (int mi = 0; mi < 2; ++mi) {
#pragma unroll
        for (int j = 0; j < 4; ++j) {
#pragma unroll
            for (int r = 0; r < 8; ++r) {
                const int g = tm + mi * 16 + r + 8 * half;   // 0..8191
                const int o = tn + j * 16 + ml;              // 0..3071
                const int s = o >> 10;                       // 0:q 1:k 2:v
                const int rem = o & 1023;
                const int h = rem >> 6;
                const int d = rem & 63;
                const int bb = g >> 11;
                const int n  = g & 2047;
                const long idx = (((long)(bb * HD + h) * ND) + n) * DD + d;
                dsts[s][idx] = f2bf(acc[mi][j][r]);
            }
        }
    }
}

// ---------------- Projection GEMM: [8192,1024] x [1024,1024]^T + bias -> fp32 ----------------
__global__ __launch_bounds__(256) void proj_gemm(const unsigned short* __restrict__ Ab,
                                                 const unsigned short* __restrict__ Wb,
                                                 const float* __restrict__ bias,
                                                 float* __restrict__ out) {
    const int lane = threadIdx.x & 31;
    const int widx = threadIdx.x >> 5;
    const int wg   = blockIdx.x * 8 + widx;
    const int mtiles = (BD * ND) / 32;            // 256
    const int tm   = (wg % mtiles) * 32;
    const int tn   = (wg / mtiles) * 64;          // 0..960
    const int half = lane >> 4;
    const int ml   = lane & 15;

    const unsigned short* arow[2];
    arow[0] = Ab + (long)(tm + ml) * KD;
    arow[1] = Ab + (long)(tm + 16 + ml) * KD;
    const unsigned short* brow[4];
#pragma unroll
    for (int j = 0; j < 4; ++j) brow[j] = Wb + (long)(tn + j * 16 + ml) * KD;

    v8f acc[2][4];
#pragma unroll
    for (int mi = 0; mi < 2; ++mi)
#pragma unroll
        for (int j = 0; j < 4; ++j) acc[mi][j] = v8f_zero();

    V16U a[2][2], b[2][4];
    auto loadA = [&](int buf, int k0) {
#pragma unroll
        for (int mi = 0; mi < 2; ++mi) {
            const unsigned short* p = arow[mi] + k0 + 8 * half;
            a[buf][mi].q[0] = *(const uint4*)(p);
            a[buf][mi].q[1] = *(const uint4*)(p + 16);
        }
    };
    auto loadB = [&](int buf, int k0) {
#pragma unroll
        for (int j = 0; j < 4; ++j) {
            const unsigned short* p = brow[j] + k0 + 8 * half;
            b[buf][j].q[0] = *(const uint4*)(p);
            b[buf][j].q[1] = *(const uint4*)(p + 16);
        }
    };

    loadA(0, 0);
    loadB(0, 0);
    for (int k0 = 0; k0 < KD; k0 += 32) {
        const int cur = (k0 >> 5) & 1;
        const int nxt = cur ^ 1;
        if (k0 + 32 < KD) { loadA(nxt, k0 + 32); loadB(nxt, k0 + 32); }
#pragma unroll
        for (int mi = 0; mi < 2; ++mi)
#pragma unroll
            for (int j = 0; j < 4; ++j)
                acc[mi][j] = __builtin_amdgcn_wmma_f32_16x16x32_bf16(
                    false, a[cur][mi].v, false, b[cur][j].v, (short)0, acc[mi][j], false, false);
    }

#pragma unroll
    for (int mi = 0; mi < 2; ++mi)
#pragma unroll
        for (int j = 0; j < 4; ++j)
#pragma unroll
            for (int r = 0; r < 8; ++r) {
                const int g = tm + mi * 16 + r + 8 * half;
                const int o = tn + j * 16 + ml;
                out[(long)g * CD + o] = acc[mi][j][r] + bias[o];
            }
}

// =====================================================================
// Flash attention: one wave per (b,h, 16-row q tile).  V tiles staged
// into LDS by the Tensor Data Mover (double buffered, s_wait_tensorcnt).
// =====================================================================
#if USE_TDM
// 2D TDM descriptor: tile 64 x 32 (2-byte elements), row stride = D,
// tensor dims {64, 2048}.  Groups per CDNA5 ISA.
// This toolchain's builtin takes 6 args: (v4u g0, v8i g1, v4i, v4i, v8i, i32 cpol).
__device__ __forceinline__ void tdm_load_v_tile(const unsigned short* gptr, unsigned lds_off) {
    const unsigned long long ga = (unsigned long long)(size_t)gptr;
    v4u g0;
    g0[0] = 1u;                                            // count=1, user mode
    g0[1] = lds_off;                                       // lds_addr
    g0[2] = (unsigned)(ga & 0xFFFFFFFFu);                  // global_addr[31:0]
    g0[3] = (unsigned)((ga >> 32) & 0x01FFFFFFu) | (2u << 30);  // addr[56:32] | type=2
    v8i g1;
    g1[0] = 0x00010000;                                    // data_size=1 (2 bytes)
    g1[1] = (int)(64u << 16);                              // tensor_dim0 = 64
    g1[2] = (int)(2048u << 16);                            // tensor_dim1 = 2048
    g1[3] = (int)(64u << 16);                              // tile_dim0  = 64
    g1[4] = 32;                                            // tile_dim1  = 32
    g1[5] = 64;                                            // tensor_dim0_stride = 64
    g1[6] = 0;
    g1[7] = 0;
    v4i z4; z4[0] = 0; z4[1] = 0; z4[2] = 0; z4[3] = 0;    // groups 2/3 unused (<=2D)
    v8i z8;
#pragma unroll
    for (int i = 0; i < 8; ++i) z8[i] = 0;
    __builtin_amdgcn_tensor_load_to_lds(g0, g1, z4, z4, z8, 0);
}
#endif

__global__ __launch_bounds__(32) void flash_attn(const unsigned short* __restrict__ Q,
                                                 const unsigned short* __restrict__ Kt,
                                                 const unsigned short* __restrict__ Vt,
                                                 unsigned short* __restrict__ O) {
    __shared__ unsigned short lds_p[16 * 32];       // P tile (C-layout staged for A-layout reads)
    __shared__ unsigned short lds_v[2][32 * 64];    // double-buffered V tiles

    const int lane = threadIdx.x;
    const int half = lane >> 4;
    const int ml   = lane & 15;
    const int qt   = blockIdx.x & 127;              // q tile within (b,h)
    const int bh   = blockIdx.x >> 7;               // b*H + h

    const unsigned short* qbase = Q  + (long)bh * ND * DD;
    const unsigned short* kbase = Kt + (long)bh * ND * DD;
    const unsigned short* vbase = Vt + (long)bh * ND * DD;

    // Q A-tiles for the two 32-wide d steps (D = 64)
    V16U aq[2];
    {
        const unsigned short* qr = qbase + (long)(qt * 16 + ml) * DD;
#pragma unroll
        for (int ks = 0; ks < 2; ++ks) {
            const unsigned short* p = qr + ks * 32 + 8 * half;
            aq[ks].q[0] = *(const uint4*)(p);
            aq[ks].q[1] = *(const uint4*)(p + 16);
        }
    }

    v8f oacc[4];
#pragma unroll
    for (int dt = 0; dt < 4; ++dt) oacc[dt] = v8f_zero();
    float mrow[8], lrow[8];
#pragma unroll
    for (int r = 0; r < 8; ++r) { mrow[r] = -1e30f; lrow[r] = 0.0f; }

    const float scale = 0.125f;                     // D^-0.5, D=64
    const int nkt = qt / 2 + 1;                     // key tiles of 32 covering keys <= qt*16+15

#if USE_TDM
    tdm_load_v_tile(vbase, (unsigned)(size_t)&lds_v[0][0]);   // prefetch kt=0
#endif

    for (int kt = 0; kt < nkt; ++kt) {
        const int key0 = kt * 32;
        const int cur = kt & 1;

#if USE_TDM
        if (kt + 1 < nkt)
            tdm_load_v_tile(vbase + (long)(key0 + 32) * DD,
                            (unsigned)(size_t)&lds_v[cur ^ 1][0]);
#else
        {   // manual staging fallback: lane copies its key row (128 B)
            const uint4* src = (const uint4*)(vbase + (long)(key0 + lane) * DD);
            uint4* dst = (uint4*)(&lds_v[cur][0] + lane * 64);
#pragma unroll
            for (int c = 0; c < 8; ++c) dst[c] = src[c];
        }
#endif

        // S = Q * K^T  (two 16-key subtiles)
        v8f s0 = v8f_zero(), s1 = v8f_zero();
#pragma unroll
        for (int j = 0; j < 2; ++j) {
            const unsigned short* kr = kbase + (long)(key0 + j * 16 + ml) * DD;
#pragma unroll
            for (int ks = 0; ks < 2; ++ks) {
                V16U bk;
                const unsigned short* p = kr + ks * 32 + 8 * half;
                bk.q[0] = *(const uint4*)(p);
                bk.q[1] = *(const uint4*)(p + 16);
                if (j == 0)
                    s0 = __builtin_amdgcn_wmma_f32_16x16x32_bf16(
                        false, aq[ks].v, false, bk.v, (short)0, s0, false, false);
                else
                    s1 = __builtin_amdgcn_wmma_f32_16x16x32_bf16(
                        false, aq[ks].v, false, bk.v, (short)0, s1, false, false);
            }
        }

        // Online softmax per row (row r+8*half lives across one 16-lane half)
        float alpha[8];
#pragma unroll
        for (int r = 0; r < 8; ++r) {
            const int qn = qt * 16 + r + 8 * half;
            float e0 = s0[r] * scale;
            float e1 = s1[r] * scale;
            if (key0 + ml > qn)      e0 = -1e30f;   // causal mask
            if (key0 + 16 + ml > qn) e1 = -1e30f;
            float mx = fmaxf(e0, e1);
#pragma unroll
            for (int msk = 1; msk < 16; msk <<= 1)
                mx = fmaxf(mx, __shfl_xor(mx, msk, 32));
            const float nm = fmaxf(mrow[r], mx);
            alpha[r] = __expf(mrow[r] - nm);
            mrow[r]  = nm;
            const float p0 = __expf(e0 - nm);
            const float p1 = __expf(e1 - nm);
            float rs = p0 + p1;
#pragma unroll
            for (int msk = 1; msk < 16; msk <<= 1)
                rs += __shfl_xor(rs, msk, 32);
            lrow[r] = lrow[r] * alpha[r] + rs;
            lds_p[(r + 8 * half) * 32 + ml]      = f2bf(p0);
            lds_p[(r + 8 * half) * 32 + 16 + ml] = f2bf(p1);
        }
        __syncthreads();   // single-wave WG: dscnt drain for lds_p (and lds_v fallback)

#if USE_TDM
        if (kt + 1 < nkt) __builtin_amdgcn_s_wait_tensorcnt(1);  // cur tile done, next in flight
        else              __builtin_amdgcn_s_wait_tensorcnt(0);
        asm volatile("" ::: "memory");
#endif

        // A-layout P from LDS: lane row m=ml, K(e) = (e<8?e:e+8)+8*half
        V16U ap;
#pragma unroll
        for (int e = 0; e < 16; ++e) {
            const int kl = (e < 8 ? e : e + 8) + 8 * half;
            ap.s[e] = lds_p[ml * 32 + kl];
        }

        // O = O*alpha + P @ V  (4 d subtiles of 16)
#pragma unroll
        for (int dt = 0; dt < 4; ++dt) {
            V16U bv;
#pragma unroll
            for (int e = 0; e < 16; ++e) {
                const int kl = (e < 8 ? e : e + 8) + 8 * half; // B-layout K index
                bv.s[e] = lds_v[cur][kl * 64 + dt * 16 + ml];
            }
            v8f c;
#pragma unroll
            for (int r = 0; r < 8; ++r) c[r] = oacc[dt][r] * alpha[r];
            oacc[dt] = __builtin_amdgcn_wmma_f32_16x16x32_bf16(
                false, ap.v, false, bv.v, (short)0, c, false, false);
        }
        __syncthreads();   // protect lds_p reuse next iteration
    }

    // Normalize and store to attn_out bf16, layout [B, N, H*D] = [8192, 1024]
    const int b = bh >> 4;
    const int h = bh & 15;
#pragma unroll
    for (int dt = 0; dt < 4; ++dt) {
#pragma unroll
        for (int r = 0; r < 8; ++r) {
            const int qn = qt * 16 + r + 8 * half;
            const float val = oacc[dt][r] / lrow[r];
            const long idx = ((long)(b * ND + qn)) * CD + h * DD + dt * 16 + ml;
            O[idx] = f2bf(val);
        }
    }
}

// ---------------- launcher ----------------
extern "C" void kernel_launch(void* const* d_in, const int* in_sizes, int n_in,
                              void* d_out, int out_size, void* d_ws, size_t ws_size,
                              hipStream_t stream) {
    const float* x      = (const float*)d_in[0];   // [B,N,C]
    const float* qkv_w  = (const float*)d_in[1];   // [3C,C]
    const float* proj_w = (const float*)d_in[2];   // [C,C]
    const float* proj_b = (const float*)d_in[3];   // [C]
    float* out = (float*)d_out;

    const size_t n_x    = (size_t)BD * ND * CD;        // 8388608
    const size_t n_qkvw = (size_t)3 * CD * CD;         // 3145728
    const size_t n_pw   = (size_t)CD * CD;             // 1048576
    const size_t n_head = (size_t)BD * HD * ND * DD;   // 8388608

    char* ws = (char*)d_ws;
    size_t off = 0;
    unsigned short* xb   = (unsigned short*)(ws + off); off += n_x    * 2;
    unsigned short* qwb  = (unsigned short*)(ws + off); off += n_qkvw * 2;
    unsigned short* pwb  = (unsigned short*)(ws + off); off += n_pw   * 2;
    unsigned short* qh   = (unsigned short*)(ws + off); off += n_head * 2;
    unsigned short* kh   = (unsigned short*)(ws + off); off += n_head * 2;
    unsigned short* vh   = (unsigned short*)(ws + off); off += n_head * 2;
    unsigned short* aout = (unsigned short*)(ws + off); off += n_x    * 2;

    // 1) bf16 conversions
    cvt_f32_bf16<<<(int)(n_x    / 256), 256, 0, stream>>>(x,      xb,  (int)n_x);
    cvt_f32_bf16<<<(int)(n_qkvw / 256), 256, 0, stream>>>(qkv_w,  qwb, (int)n_qkvw);
    cvt_f32_bf16<<<(int)(n_pw   / 256), 256, 0, stream>>>(proj_w, pwb, (int)n_pw);

    // 2) QKV GEMM: waves = 256 * 48 = 12288 -> 1536 blocks x 8 waves
    qkv_gemm<<<1536, 256, 0, stream>>>(xb, qwb, qh, kh, vh);

    // 3) Flash attention: B*H*(N/16) = 8192 single-wave blocks
    flash_attn<<<8192, 32, 0, stream>>>(qh, kh, vh, aout);

    // 4) Projection GEMM: waves = 256 * 16 = 4096 -> 512 blocks x 8 waves
    proj_gemm<<<512, 256, 0, stream>>>(aout, pwb, proj_b, out);
}
